// BinLinear_45887430590600
// MI455X (gfx1250) — 8-bit matrix operand order repaired and checked by reading, compiled, not run on hardware
//
#include <hip/hip_runtime.h>
#include <hip/hip_bf16.h>
#include <stdint.h>

#define B_DIM   8192
#define IN_DIM  4096
#define OUT_DIM 4096
#define EPSV    1e-4f

#define KTILES  (IN_DIM / 128)

typedef __attribute__((ext_vector_type(16))) int          v16i;
typedef __attribute__((ext_vector_type(8)))  float        v8f;
typedef __attribute__((ext_vector_type(4)))  unsigned int v4u;
typedef __attribute__((ext_vector_type(4)))  float        v4f;

__device__ __forceinline__ unsigned fp8_sign(float v) {
    return v > 0.f ? 0x38u : (v < 0.f ? 0xB8u : 0x00u);
}

#define WS_SUM    ((size_t)0)
#define WS_SUMSQ  ((size_t)IN_DIM * 4)
#define WS_MU     ((size_t)IN_DIM * 8)
#define WS_RSTD   ((size_t)IN_DIM * 12)
#define WS_SCALE  ((size_t)IN_DIM * 16)
#define WS_XBIN   ((size_t)IN_DIM * 20)
#define WS_WBIN   (WS_XBIN + (size_t)B_DIM * IN_DIM)

__global__ void k_zero(float* p) {
    p[blockIdx.x * 256 + threadIdx.x] = 0.f;
}

#define ROW_SPLIT 64
__global__ __launch_bounds__(256) void k_col_stats(const float* __restrict__ x,
                                                   float* __restrict__ sum,
                                                   float* __restrict__ sumsq) {
    const int col = blockIdx.x * 256 + threadIdx.x;
    const int r0  = blockIdx.y * (B_DIM / ROW_SPLIT);
    const float* p = x + (size_t)r0 * IN_DIM + col;
    float s = 0.f, s2 = 0.f;
    #pragma unroll 4
    for (int r = 0; r < B_DIM / ROW_SPLIT; ++r) {
        float v = p[(size_t)r * IN_DIM];
        s += v;
        s2 += v * v;
    }
    atomicAdd(&sum[col], s);
    atomicAdd(&sumsq[col], s2);
}

__global__ void k_finalize(const float* __restrict__ sum, const float* __restrict__ sumsq,
                           float* __restrict__ mu, float* __restrict__ rstd) {
    const int c = blockIdx.x * 256 + threadIdx.x;
    const float m = sum[c] * (1.f / B_DIM);
    const float v = sumsq[c] * (1.f / B_DIM) - m * m;
    mu[c]   = m;
    rstd[c] = rsqrtf(v + EPSV);
}

__global__ __launch_bounds__(256) void k_bin_x(const float* __restrict__ x,
                                               const float* __restrict__ mu,
                                               const float* __restrict__ rstd,
                                               const float* __restrict__ gamma,
                                               const float* __restrict__ beta,
                                               int8_t* __restrict__ xbin) {
    const int idx = blockIdx.x * 256 + threadIdx.x;
    const int row = idx / (IN_DIM / 8);
    const int k   = (idx % (IN_DIM / 8)) * 8;
    const int c4  = k >> 2;

    const v4f x0 = ((const v4f*)x)[idx * 2];
    const v4f x1 = ((const v4f*)x)[idx * 2 + 1];
    const v4f m0 = ((const v4f*)mu)[c4],    m1 = ((const v4f*)mu)[c4 + 1];
    const v4f r0 = ((const v4f*)rstd)[c4],  r1 = ((const v4f*)rstd)[c4 + 1];
    const v4f g0 = ((const v4f*)gamma)[c4], g1 = ((const v4f*)gamma)[c4 + 1];
    const v4f b0 = ((const v4f*)beta)[c4],  b1 = ((const v4f*)beta)[c4 + 1];
    const v4f n0 = (x0 - m0) * r0 * g0 + b0;
    const v4f n1 = (x1 - m1) * r1 * g1 + b1;

    uint64_t packed =  (uint64_t)fp8_sign(n0.x)
                    | ((uint64_t)fp8_sign(n0.y) << 8)
                    | ((uint64_t)fp8_sign(n0.z) << 16)
                    | ((uint64_t)fp8_sign(n0.w) << 24)
                    | ((uint64_t)fp8_sign(n1.x) << 32)
                    | ((uint64_t)fp8_sign(n1.y) << 40)
                    | ((uint64_t)fp8_sign(n1.z) << 48)
                    | ((uint64_t)fp8_sign(n1.w) << 56);

    const size_t base = ((size_t)(row >> 4) * KTILES + (k >> 7)) * 2048;
    const int lane    = ((k >> 4) & 1) * 16 + (row & 15);
    const int off     = ((k >> 5) & 3) * 16 + ((k >> 3) & 1) * 8;
    *(uint64_t*)(xbin + base + lane * 64 + off) = packed;
}

__global__ __launch_bounds__(256) void k_bin_w(const float* __restrict__ w,
                                               int8_t* __restrict__ wbin,
                                               float* __restrict__ scale) {
    __shared__ float red[256];
    const int n = blockIdx.x;
    const int t = threadIdx.x;
    const int k = t * 16;
    const float* row = w + (size_t)n * IN_DIM + k;

    float vals[16];
    float s = 0.f;
    #pragma unroll
    for (int i = 0; i < 16; ++i) { vals[i] = row[i]; s += vals[i]; }
    red[t] = s;
    __syncthreads();
    for (int off = 128; off > 0; off >>= 1) {
        if (t < off) red[t] += red[t + off];
        __syncthreads();
    }
    const float mean = red[0] * (1.f / IN_DIM);
    __syncthreads();

    float sa = 0.f;
    unsigned pk[4] = {0u, 0u, 0u, 0u};
    #pragma unroll
    for (int i = 0; i < 16; ++i) {
        float wc = vals[i] - mean;
        wc = fminf(1.f, fmaxf(-1.f, wc));
        sa += fabsf(wc);
        pk[i >> 2] |= fp8_sign(wc) << ((i & 3) * 8);
    }
    const size_t base = ((size_t)(n >> 4) * KTILES + (k >> 7)) * 2048;
    const int lane    = ((k >> 4) & 1) * 16 + (n & 15);
    const int off     = ((k >> 5) & 3) * 16;
    *(v4u*)(wbin + base + lane * 64 + off) = (v4u){pk[0], pk[1], pk[2], pk[3]};

    red[t] = sa;
    __syncthreads();
    for (int off2 = 128; off2 > 0; off2 >>= 1) {
        if (t < off2) red[t] += red[t + off2];
        __syncthreads();
    }
    if (t == 0) scale[n] = red[0] * (1.f / IN_DIM);
}

__device__ __forceinline__ v16i load_frag(const int8_t* p) {
    union { v16i v; v4u u[4]; } f;
    f.u[0] = *(const v4u*)(p);
    f.u[1] = *(const v4u*)(p + 16);
    f.u[2] = *(const v4u*)(p + 32);
    f.u[3] = *(const v4u*)(p + 48);
    return f.v;
}

__global__ __launch_bounds__(256) void k_bin_gemm(const int8_t* __restrict__ xbin,
                                                  const int8_t* __restrict__ wbin,
                                                  const float* __restrict__ bias,
                                                  const float* __restrict__ scale,
                                                  float* __restrict__ out) {
    const int lane   = threadIdx.x & 31;
    const int wave   = threadIdx.x >> 5;
    const int wm     = wave >> 1;
    const int wn     = wave & 1;
    const int laneLo = lane & 15;
    const int laneHi = lane >> 4;

    const int m0 = blockIdx.y * 256 + wm * 64;
    const int n0 = blockIdx.x * 128 + wn * 64;

    v8f acc[4][4];
    #pragma unroll
    for (int mi = 0; mi < 4; ++mi)
        #pragma unroll
        for (int ni = 0; ni < 4; ++ni)
            acc[mi][ni] = (v8f){0.f, 0.f, 0.f, 0.f, 0.f, 0.f, 0.f, 0.f};

    const int8_t* aBase[4];
    #pragma unroll
    for (int mi = 0; mi < 4; ++mi)
        aBase[mi] = xbin + ((size_t)(m0 / 16 + mi) * KTILES) * 2048 + lane * 64;
    const int8_t* bBase[4];
    #pragma unroll
    for (int ni = 0; ni < 4; ++ni)
        bBase[ni] = wbin + ((size_t)(n0 / 16 + ni) * KTILES) * 2048 + lane * 64;

    for (int kt = 0; kt < KTILES; ++kt) {
        const size_t koff = (size_t)kt * 2048;
        v16i a[4], b[4];
        #pragma unroll
        for (int mi = 0; mi < 4; ++mi) a[mi] = load_frag(aBase[mi] + koff);
        #pragma unroll
        for (int ni = 0; ni < 4; ++ni) b[ni] = load_frag(bBase[ni] + koff);

        #pragma unroll
        for (int mi = 0; mi < 4; ++mi)
            #pragma unroll
            for (int ni = 0; ni < 4; ++ni)
                acc[mi][ni] = __builtin_amdgcn_wmma_f32_16x16x128_fp8_fp8(
                    a[mi], b[ni], (short)0, acc[mi][ni],
                     false,  false);
    }

    #pragma unroll
    for (int ni = 0; ni < 4; ++ni) {
        const int col = n0 + ni * 16 + laneLo;
        const float sc = scale[col];
        const float bs = bias[col];
        #pragma unroll
        for (int mi = 0; mi < 4; ++mi) {
            #pragma unroll
            for (int e = 0; e < 8; ++e) {
                const int row = m0 + mi * 16 + laneHi * 8 + e;
                const float y = (acc[mi][ni][e] + bs) * sc;
                out[(size_t)row * OUT_DIM + col] = fmaxf(y, 0.f);
            }
        }
    }
}

extern "C" void kernel_launch(void* const* d_in, const int* in_sizes, int n_in,
                              void* d_out, int out_size, void* d_ws, size_t ws_size,
                              hipStream_t stream) {
    const float* x      = (const float*)d_in[0];
    const float* gamma  = (const float*)d_in[1];
    const float* beta   = (const float*)d_in[2];
    const float* weight = (const float*)d_in[3];
    const float* bias   = (const float*)d_in[4];
    float* out = (float*)d_out;

    char* ws = (char*)d_ws;
    float*  sum   = (float*)(ws + WS_SUM);
    float*  sumsq = (float*)(ws + WS_SUMSQ);
    float*  mu    = (float*)(ws + WS_MU);
    float*  rstd  = (float*)(ws + WS_RSTD);
    float*  scale = (float*)(ws + WS_SCALE);
    int8_t* xbin  = (int8_t*)(ws + WS_XBIN);
    int8_t* wbin  = (int8_t*)(ws + WS_WBIN);

    k_zero<<<(2 * IN_DIM) / 256, 256, 0, stream>>>(sum);
    k_col_stats<<<dim3(IN_DIM / 256, ROW_SPLIT), 256, 0, stream>>>(x, sum, sumsq);
    k_finalize<<<IN_DIM / 256, 256, 0, stream>>>(sum, sumsq, mu, rstd);
    k_bin_x<<<((size_t)B_DIM * IN_DIM / 8) / 256, 256, 0, stream>>>(x, mu, rstd, gamma, beta, xbin);
    k_bin_w<<<OUT_DIM, 256, 0, stream>>>(weight, wbin, scale);
    k_bin_gemm<<<dim3(OUT_DIM / 128, B_DIM / 256), 256, 0, stream>>>(xbin, wbin, bias, scale, out);
}
